// Attention_8272107012675
// MI455X (gfx1250) — matrix operand order repaired and checked by reading, not compiled, not run on hardware
//
#include <hip/hip_runtime.h>
#include <hip/hip_bf16.h>

// ---------------------------------------------------------------------------
// Problem constants
// ---------------------------------------------------------------------------
#define BATCH 32
#define SEQ   1025          // real sequence length (HW + cls)
#define SEQP  1040          // padded to 65 tiles of 16
#define NT    65            // SEQP / 16
#define DIM   768
#define KSTEPS (DIM / 32)   // 24 wmma k-steps of 32
#define WELEM (DIM * DIM)   // 589824
#define SCALE 0.03608439182435161f   // 1/sqrt(768)

#define MFLAT (BATCH * SEQP)         // 33280 flat rows (batch-contiguous)
#define WPB   8                      // waves per block in diag kernel
#define UGB   ((NT + WPB - 1) / WPB) // 9 u-groups per batch
#define QROWS (WPB * 16)             // 128 staged q rows
#define KTILE_BYTES (16 * DIM * 2)   // 24576
#define QBYTES (QROWS * DIM * 2)     // 196608
#define DIAG_SMEM (QBYTES + 2 * KTILE_BYTES)  // 245760 < 320KB WGP LDS

typedef _Float16 h16v __attribute__((ext_vector_type(16)));
typedef _Float16 h8v  __attribute__((ext_vector_type(8)));
typedef _Float16 h4v  __attribute__((ext_vector_type(4)));
typedef float    f8v  __attribute__((ext_vector_type(8)));

// ---------------------------------------------------------------------------
// WMMA fragment loaders (per CDNA5 ISA 7.12.2 VGPR layouts, wave32)
//   A 16x32 f16 : lane L holds row M=L&15; lanes<16: K=0..7 & 16..23,
//                 lanes>=16: K=8..15 & 24..31 (2 halves per VGPR)
//   B 32x16 f16 : lane L holds col N=L&15; lanes<16: K=0..7 & 16..23,
//                 lanes>=16: K=8..15 & 24..31 (same order as A)
//   C/D 16x16 f32: lane L col N=L&15, VGPR r -> M = r + 8*(L>>4)
// ---------------------------------------------------------------------------
__device__ __forceinline__ h16v load_a_frag(const _Float16* rowPtr, int hi, int kb) {
  h8v lo = *(const h8v*)(rowPtr + kb + hi * 8);
  h8v up = *(const h8v*)(rowPtr + kb + 16 + hi * 8);
  h16v r;
#pragma unroll
  for (int i = 0; i < 8; ++i) { r[i] = lo[i]; r[8 + i] = up[i]; }
  return r;
}

__device__ __forceinline__ h16v load_b_frag(const _Float16* rowPtr, int hi, int kb) {
  return load_a_frag(rowPtr, hi, kb);
}

__device__ __forceinline__ f8v wmma_f16(h16v a, h16v b, f8v c) {
  return __builtin_amdgcn_wmma_f32_16x16x32_f16(false, a, false, b,
                                                (short)0, c, false, false);
}

__device__ __forceinline__ float rowmax16(float v) {
#pragma unroll
  for (int m = 1; m < 16; m <<= 1) v = fmaxf(v, __shfl_xor(v, m, 16));
  return v;
}
__device__ __forceinline__ float rowsum16(float v) {
#pragma unroll
  for (int m = 1; m < 16; m <<= 1) v += __shfl_xor(v, m, 16);
  return v;
}

// CDNA5 async global->LDS copy, 16B per lane, tracked by ASYNCcnt (in-order)
__device__ __forceinline__ void async_cp16(unsigned ldsOff, const void* gaddr) {
  asm volatile("global_load_async_to_lds_b128 %0, %1, off"
               : : "v"(ldsOff), "v"(gaddr) : "memory");
}

// ---------------------------------------------------------------------------
// Kernel 1: build encoded (fp32 -> d_out second half) + f16 padded copy in ws
// ---------------------------------------------------------------------------
__global__ void encode_kernel(const float* __restrict__ x,
                              const float* __restrict__ cw,
                              const float* __restrict__ cb,
                              const float* __restrict__ cls,
                              const float* __restrict__ pos,
                              _Float16* __restrict__ encH,
                              float* __restrict__ outEnc) {
  size_t idx = (size_t)blockIdx.x * blockDim.x + threadIdx.x;
  const size_t total = (size_t)BATCH * SEQP * (DIM / 4);
  if (idx >= total) return;
  int d  = (int)(idx % (DIM / 4)) * 4;
  size_t rest = idx / (DIM / 4);
  int s = (int)(rest % SEQP);
  int b = (int)(rest / SEQP);

  float v0 = 0.f, v1 = 0.f, v2 = 0.f, v3 = 0.f;
  if (s == 0) {
    v0 = cls[d + 0] + pos[d + 0];
    v1 = cls[d + 1] + pos[d + 1];
    v2 = cls[d + 2] + pos[d + 2];
    v3 = cls[d + 3] + pos[d + 3];
  } else if (s < SEQ) {
    float p = x[(size_t)b * 1024 + (s - 1)];
    const float* pr = pos + (size_t)s * DIM + d;
    v0 = fmaf(p, cw[d + 0], cb[d + 0]) + pr[0];
    v1 = fmaf(p, cw[d + 1], cb[d + 1]) + pr[1];
    v2 = fmaf(p, cw[d + 2], cb[d + 2]) + pr[2];
    v3 = fmaf(p, cw[d + 3], cb[d + 3]) + pr[3];
  }
  h4v h = { (_Float16)v0, (_Float16)v1, (_Float16)v2, (_Float16)v3 };
  *(h4v*)(encH + ((size_t)b * SEQP + s) * DIM + d) = h;
  if (s < SEQ) {
    float4 f = make_float4(v0, v1, v2, v3);
    *(float4*)(outEnc + ((size_t)b * SEQ + s) * DIM + d) = f;
  }
}

// ---------------------------------------------------------------------------
// Kernel 2: convert Wq/Wk/Wv to f16
// ---------------------------------------------------------------------------
__global__ void cvt_w_kernel(const float* __restrict__ Wq,
                             const float* __restrict__ Wk,
                             const float* __restrict__ Wv,
                             _Float16* __restrict__ WqH,
                             _Float16* __restrict__ WkH,
                             _Float16* __restrict__ WvH) {
  int i = blockIdx.x * blockDim.x + threadIdx.x;
  if (i < WELEM) {
    WqH[i] = (_Float16)Wq[i];
    WkH[i] = (_Float16)Wk[i];
    WvH[i] = (_Float16)Wv[i];
  }
}

// ---------------------------------------------------------------------------
// Kernel 3: q / k projection GEMM  (C = encH @ W^T), f16 out.
//   Flat M = 33280 rows. 32x32 tile per wave (4 accumulators, A/B reused 2x).
// ---------------------------------------------------------------------------
__global__ __launch_bounds__(128) void qk_gemm_kernel(
    const _Float16* __restrict__ encH,
    const _Float16* __restrict__ WqH, const _Float16* __restrict__ WkH,
    _Float16* __restrict__ qH, _Float16* __restrict__ kH) {
  const int lane = threadIdx.x & 31, wave = threadIdx.x >> 5;
  const int nIdx = lane & 15, hi = lane >> 4;
  const int n0 = (blockIdx.x * 4 + wave) * 32;
  const int m0 = blockIdx.y * 32;

  const _Float16* W   = blockIdx.z ? WkH : WqH;
  _Float16*       dst = blockIdx.z ? kH  : qH;

  const _Float16* arow0 = encH + (size_t)(m0 + nIdx) * DIM;
  const _Float16* arow1 = arow0 + (size_t)16 * DIM;
  const _Float16* brow0 = W + (size_t)(n0 + nIdx) * DIM;
  const _Float16* brow1 = brow0 + (size_t)16 * DIM;

  f8v a00 = {}, a01 = {}, a10 = {}, a11 = {};
#pragma unroll
  for (int kk = 0; kk < KSTEPS; ++kk) {
    const int kb = kk * 32;
    h16v fa0 = load_a_frag(arow0, hi, kb);
    h16v fa1 = load_a_frag(arow1, hi, kb);
    h16v fb0 = load_b_frag(brow0, hi, kb);
    h16v fb1 = load_b_frag(brow1, hi, kb);
    a00 = wmma_f16(fa0, fb0, a00);
    a01 = wmma_f16(fa0, fb1, a01);
    a10 = wmma_f16(fa1, fb0, a10);
    a11 = wmma_f16(fa1, fb1, a11);
  }

  _Float16* out = dst + (size_t)m0 * DIM + n0 + nIdx;
#pragma unroll
  for (int r = 0; r < 8; ++r) {
    const size_t ro = (size_t)(r + 8 * hi) * DIM;
    out[ro]                         = (_Float16)a00[r];
    out[ro + 16]                    = (_Float16)a01[r];
    out[ro + (size_t)16 * DIM]      = (_Float16)a10[r];
    out[ro + (size_t)16 * DIM + 16] = (_Float16)a11[r];
  }
}

// ---------------------------------------------------------------------------
// Kernel 4: streaming diag-of-softmax, LDS-resident q, async double-buffered k
// ---------------------------------------------------------------------------
__global__ __launch_bounds__(256) void diag_softmax_kernel(
    const _Float16* __restrict__ qH, const _Float16* __restrict__ kH,
    float* __restrict__ diag) {
  extern __shared__ __align__(16) _Float16 smem[];
  _Float16* qLds = smem;                        // QROWS x DIM
  _Float16* kLds = smem + (size_t)QROWS * DIM;  // 2 x (16 x DIM)

  const int tid  = threadIdx.x;
  const int lane = tid & 31, wave = tid >> 5;
  const int nIdx = lane & 15, hi = lane >> 4;
  const int b = blockIdx.x / UGB, ugrp = blockIdx.x % UGB;

  int rowbase = ugrp * QROWS;
  if (rowbase > SEQP - QROWS) rowbase = SEQP - QROWS;   // clamp ragged tail
  int ut = ugrp * WPB + wave;
  if (ut > NT - 1) ut = NT - 1;                         // clamped waves dup work
  const int ldsRow = ut * 16 - rowbase;                 // >= 0 by construction

  const unsigned qDst = (unsigned)(uintptr_t)qLds;
  const unsigned kDst = (unsigned)(uintptr_t)kLds;

  // stage 128 q rows into LDS (48 x 16B chunks per thread, async)
  {
    const char* qSrc = (const char*)(qH + ((size_t)b * SEQP + rowbase) * DIM);
#pragma unroll
    for (int t = 0; t < QBYTES / (256 * 16); ++t) {
      const int c = tid + t * 256;
      async_cp16(qDst + c * 16, qSrc + (size_t)c * 16);
    }
  }
  // k tile 0 into buffer 0 (6 x 16B chunks per thread)
  {
    const char* src = (const char*)(kH + (size_t)b * SEQP * DIM);
#pragma unroll
    for (int t = 0; t < KTILE_BYTES / (256 * 16); ++t) {
      const int c = tid + t * 256;
      async_cp16(kDst + c * 16, src + (size_t)c * 16);
    }
  }

  const _Float16* qrow = qLds + (size_t)(ldsRow + nIdx) * DIM;

  float m[8], l[8], ds[8];
#pragma unroll
  for (int r = 0; r < 8; ++r) {
    m[r] = -__builtin_inff(); l[r] = 0.f; ds[r] = -__builtin_inff();
  }

  for (int vt = 0; vt < NT; ++vt) {
    if (vt + 1 < NT) {
      const char* src =
          (const char*)(kH + ((size_t)b * SEQP + (size_t)(vt + 1) * 16) * DIM);
      const unsigned dst = kDst + ((vt + 1) & 1) * KTILE_BYTES;
#pragma unroll
      for (int t = 0; t < KTILE_BYTES / (256 * 16); ++t) {
        const int c = tid + t * 256;
        async_cp16(dst + c * 16, src + (size_t)c * 16);
      }
      // ASYNCcnt completes in order: <=6 outstanding => older tiles landed
      asm volatile("s_wait_asynccnt 0x6" ::: "memory");
    } else {
      asm volatile("s_wait_asynccnt 0x0" ::: "memory");
    }
    __syncthreads();   // tile vt visible to all waves

    const _Float16* krow = kLds + (size_t)(vt & 1) * (16 * DIM) +
                           (size_t)nIdx * DIM;
    f8v acc = {};
#pragma unroll
    for (int kk = 0; kk < KSTEPS; ++kk)
      acc = wmma_f16(load_a_frag(qrow, hi, kk * 32),
                     load_b_frag(krow, hi, kk * 32), acc);

    const bool colValid = (vt * 16 + nIdx) < SEQ;
#pragma unroll
    for (int r = 0; r < 8; ++r) {
      float sv = colValid ? acc[r] * SCALE : -__builtin_inff();
      float tmax = rowmax16(sv);
      float mn = fmaxf(m[r], tmax);
      float p = __expf(sv - mn);            // exp(-inf)==0 masks padded cols
      float ps = rowsum16(p);
      l[r] = l[r] * __expf(m[r] - mn) + ps;
      m[r] = mn;
      if (vt == ut && nIdx == r + 8 * hi) ds[r] = sv;   // diagonal M==N
    }
    __syncthreads();   // all waves done reading buf before it is overwritten
  }

  // owner lanes (N == M) hold the diagonal score; stats are lane-replicated
#pragma unroll
  for (int r = 0; r < 8; ++r) {
    if (nIdx == r + 8 * hi) {
      const int s = ut * 16 + r + 8 * hi;
      if (s < SEQ)
        diag[(size_t)b * SEQ + s] = __expf(ds[r] - m[r]) / l[r];
    }
  }
}

// ---------------------------------------------------------------------------
// Kernel 5: V GEMM (32x32 wave tiles, flat M) fused with diag scaling, f32 out
// ---------------------------------------------------------------------------
__global__ __launch_bounds__(128) void v_gemm_diag_kernel(
    const _Float16* __restrict__ encH, const _Float16* __restrict__ WvH,
    const float* __restrict__ diag, float* __restrict__ out) {
  const int lane = threadIdx.x & 31, wave = threadIdx.x >> 5;
  const int nIdx = lane & 15, hi = lane >> 4;
  const int n0 = (blockIdx.x * 4 + wave) * 32;
  const int m0 = blockIdx.y * 32;

  const _Float16* arow0 = encH + (size_t)(m0 + nIdx) * DIM;
  const _Float16* arow1 = arow0 + (size_t)16 * DIM;
  const _Float16* brow0 = WvH + (size_t)(n0 + nIdx) * DIM;
  const _Float16* brow1 = brow0 + (size_t)16 * DIM;

  f8v a00 = {}, a01 = {}, a10 = {}, a11 = {};
#pragma unroll
  for (int kk = 0; kk < KSTEPS; ++kk) {
    const int kb = kk * 32;
    h16v fa0 = load_a_frag(arow0, hi, kb);
    h16v fa1 = load_a_frag(arow1, hi, kb);
    h16v fb0 = load_b_frag(brow0, hi, kb);
    h16v fb1 = load_b_frag(brow1, hi, kb);
    a00 = wmma_f16(fa0, fb0, a00);
    a01 = wmma_f16(fa0, fb1, a01);
    a10 = wmma_f16(fa1, fb0, a10);
    a11 = wmma_f16(fa1, fb1, a11);
  }

#pragma unroll
  for (int r = 0; r < 8; ++r) {
    // m-subtile 0
    {
      const int mrow = m0 + r + 8 * hi;
      const int bb = mrow / SEQP, s = mrow % SEQP;
      if (s < SEQ) {
        const float dg = diag[(size_t)bb * SEQ + s];
        float* o = out + ((size_t)bb * SEQ + s) * DIM + n0 + nIdx;
        o[0]  = a00[r] * dg;
        o[16] = a01[r] * dg;
      }
    }
    // m-subtile 1
    {
      const int mrow = m0 + 16 + r + 8 * hi;
      const int bb = mrow / SEQP, s = mrow % SEQP;
      if (s < SEQ) {
        const float dg = diag[(size_t)bb * SEQ + s];
        float* o = out + ((size_t)bb * SEQ + s) * DIM + n0 + nIdx;
        o[0]  = a10[r] * dg;
        o[16] = a11[r] * dg;
      }
    }
  }
}

// ---------------------------------------------------------------------------
// Launch
// ---------------------------------------------------------------------------
extern "C" void kernel_launch(void* const* d_in, const int* in_sizes, int n_in,
                              void* d_out, int out_size, void* d_ws, size_t ws_size,
                              hipStream_t stream) {
  (void)in_sizes; (void)n_in; (void)out_size; (void)ws_size;
  const float* x   = (const float*)d_in[0];
  const float* cw  = (const float*)d_in[1];
  const float* cb  = (const float*)d_in[2];
  const float* cls = (const float*)d_in[3];
  const float* pos = (const float*)d_in[4];
  const float* Wq  = (const float*)d_in[5];
  const float* Wk  = (const float*)d_in[6];
  const float* Wv  = (const float*)d_in[7];

  float* out_attn = (float*)d_out;                              // [32,1025,768]
  float* out_enc  = out_attn + (size_t)BATCH * SEQ * DIM;       // [32,1025,768]

  const size_t EP = (size_t)BATCH * SEQP * DIM;                 // padded f16 plane
  _Float16* encH = (_Float16*)d_ws;
  _Float16* qH   = encH + EP;
  _Float16* kH   = qH + EP;
  _Float16* WqH  = kH + EP;
  _Float16* WkH  = WqH + WELEM;
  _Float16* WvH  = WkH + WELEM;
  float*    diag = (float*)(WvH + WELEM);                       // [32,1025]

  {
    const size_t total = (size_t)BATCH * SEQP * (DIM / 4);
    encode_kernel<<<(unsigned)((total + 255) / 256), 256, 0, stream>>>(
        x, cw, cb, cls, pos, encH, out_enc);
  }
  cvt_w_kernel<<<(WELEM + 255) / 256, 256, 0, stream>>>(Wq, Wk, Wv, WqH, WkH, WvH);

  qk_gemm_kernel<<<dim3(6, MFLAT / 32, 2), 128, 0, stream>>>(
      encH, WqH, WkH, qH, kH);

  diag_softmax_kernel<<<BATCH * UGB, 256, DIAG_SMEM, stream>>>(qH, kH, diag);

  v_gemm_diag_kernel<<<dim3(6, MFLAT / 32), 128, 0, stream>>>(
      encH, WvH, diag, out_attn);
}
